// BasicBlock_54357106098411
// MI455X (gfx1250) — 8-bit matrix operand order repaired and checked by reading, compiled, not run on hardware
//
#include <hip/hip_runtime.h>
#include <hip/hip_bf16.h>
#include <stdint.h>

typedef int v8i __attribute__((ext_vector_type(8)));

#define NB   32
#define CCH  64
#define HH   112
#define WW   112
#define HP   114
#define WP   114
#define NHW  (NB * HH * WW)
#define KTOT 576

#define WS_SCALE 0
#define WS_SUM   256
#define WS_SQ    512
#define WS_KC    768
#define WS_DC    1024
#define WS_BFRAG 4096
#define WS_ABIN  65536
#define ABIN_BYTES ((size_t)NB * HP * WP * CCH)

__global__ void prep_scale_stats(const float* __restrict__ w,
                                 float* __restrict__ scale,
                                 float* __restrict__ gsum,
                                 float* __restrict__ gsq) {
    int oc = threadIdx.x;
    const float* wp = w + oc * KTOT;
    float s = 0.f;
    for (int j = 0; j < KTOT; ++j) s += fabsf(wp[j]);
    scale[oc] = s * (1.0f / (float)KTOT);
    gsum[oc] = 0.f;
    gsq[oc]  = 0.f;
}

__global__ void prep_bfrag(const float* __restrict__ w, int* __restrict__ bfrag) {
    int tg   = blockIdx.x;
    int tap  = tg >> 2;
    int g    = tg & 3;
    int lane = threadIdx.x;
    int oc   = g * 16 + (lane & 15);
    int khi  = (lane >> 4) & 1;
    int base = (tg * 32 + lane) * 8;
    for (int v = 0; v < 8; ++v) {
        int kb = (v >> 1) * 16 + khi * 8 + (v & 1) * 4;
        unsigned int word = 0;
        for (int b = 0; b < 4; ++b) {
            int ic = kb + b;
            float wv = w[oc * KTOT + ic * 9 + tap];
            unsigned int sb = (wv > 0.f) ? 0x01u : 0xFFu;
            word |= sb << (8 * b);
        }
        bfrag[base + v] = (int)word;
    }
}

__global__ void __launch_bounds__(256)
binarize_act(const float* __restrict__ x, const float* __restrict__ b0,
             unsigned char* __restrict__ abin) {
    __shared__ __align__(16) unsigned char tile[WW * CCH];
    int nh  = blockIdx.x;
    int tid = threadIdx.x;
    int n = nh / HH, h = nh % HH;
    size_t xbase = (size_t)n * (CCH * HH * WW) + (size_t)h * WW;
#pragma unroll
    for (int it = 0; it < 28; ++it) {
        int idx = tid + it * 256;
        int c = idx / WW, wp = idx % WW;
        float v = x[xbase + (size_t)c * (HH * WW) + wp] + b0[c];
        signed char s = (signed char)((v > 0.f) - (v < 0.f));
        tile[wp * CCH + c] = (unsigned char)s;
    }
    __syncthreads();
    const unsigned int* tw = (const unsigned int*)tile;
    unsigned int* ow = (unsigned int*)abin +
                       (((size_t)n * HP + (h + 1)) * WP + 1) * 16;
#pragma unroll
    for (int it = 0; it < 7; ++it) {
        int i = tid + it * 256;
        ow[i] = tw[i];
    }
}

__device__ __forceinline__ v8i load_afrag(const unsigned char* __restrict__ p,
                                          const int off) {
    unsigned long long u0 = *(const unsigned long long*)(p + off);
    unsigned long long u1 = *(const unsigned long long*)(p + off + 16);
    unsigned long long u2 = *(const unsigned long long*)(p + off + 32);
    unsigned long long u3 = *(const unsigned long long*)(p + off + 48);
    v8i a;
    a[0] = (int)(unsigned)u0; a[1] = (int)(unsigned)(u0 >> 32);
    a[2] = (int)(unsigned)u1; a[3] = (int)(unsigned)(u1 >> 32);
    a[4] = (int)(unsigned)u2; a[5] = (int)(unsigned)(u2 >> 32);
    a[6] = (int)(unsigned)u3; a[7] = (int)(unsigned)(u3 >> 32);
    return a;
}

__device__ __forceinline__ void conv_tile2(const unsigned char* __restrict__ abin,
                                           const int* ldsB,
                                           int n0, int h0, int wb0,
                                           int n1, int h1, int wb1,
                                           int lane, v8i acc[8]) {
    const int m = lane & 15;
    const int half8 = (lane >> 4) & 1;
    const unsigned char* p0 =
        abin + (((size_t)n0 * HP + h0) * WP + (wb0 + m)) * CCH + half8 * 8;
    const unsigned char* p1 =
        abin + (((size_t)n1 * HP + h1) * WP + (wb1 + m)) * CCH + half8 * 8;
    const v8i zero = {0, 0, 0, 0, 0, 0, 0, 0};
#pragma unroll
    for (int g = 0; g < 8; ++g) acc[g] = zero;
#pragma unroll
    for (int kh = 0; kh < 3; ++kh) {
#pragma unroll
        for (int kw = 0; kw < 3; ++kw) {
            const int off = (kh * WP + kw) * CCH;
            v8i a0 = load_afrag(p0, off);
            v8i a1 = load_afrag(p1, off);
            const int tap = kh * 3 + kw;
#pragma unroll
            for (int g = 0; g < 4; ++g) {
                v8i b = *(const v8i*)(ldsB + ((tap * 4 + g) * 32 + lane) * 8);
                acc[g] = __builtin_amdgcn_wmma_i32_16x16x64_iu8(
                    true, a0, true, b, acc[g], false, false);
                acc[4 + g] = __builtin_amdgcn_wmma_i32_16x16x64_iu8(
                    true, a1, true, b, acc[4 + g], false, false);
            }
        }
    }
}

__device__ __forceinline__ void decode_tile(int tile, int& n, int& h, int& wb) {
    int tx = tile % 7, nh = tile / 7;
    h = nh % HH; n = nh / HH; wb = tx * 16;
}

__global__ void __launch_bounds__(256)
conv_stats(const unsigned char* __restrict__ abin, const int* __restrict__ bfragG,
           const float* __restrict__ scale,
           float* __restrict__ gsum, float* __restrict__ gsq) {
    __shared__ __align__(64) int ldsB[9216];
    __shared__ float ssum[CCH], ssq[CCH];
    int tid = threadIdx.x;
#pragma unroll
    for (int it = 0; it < 36; ++it) ldsB[tid + it * 256] = bfragG[tid + it * 256];
    if (tid < CCH) { ssum[tid] = 0.f; ssq[tid] = 0.f; }
    __syncthreads();

    int wave = tid >> 5, lane = tid & 31;
    int pair = blockIdx.x * 8 + wave;
    int n0, h0, wb0, n1, h1, wb1;
    decode_tile(pair * 2, n0, h0, wb0);
    decode_tile(pair * 2 + 1, n1, h1, wb1);

    v8i acc[8];
    conv_tile2(abin, ldsB, n0, h0, wb0, n1, h1, wb1, lane, acc);

#pragma unroll
    for (int g = 0; g < 4; ++g) {
        int oc = g * 16 + (lane & 15);
        float fs = 0.f, fq = 0.f;
#pragma unroll
        for (int r = 0; r < 8; ++r) {
            float v0 = (float)acc[g][r];
            float v1 = (float)acc[4 + g][r];
            fs += v0 + v1;
            fq += v0 * v0 + v1 * v1;
        }
        float sc = scale[oc];
        atomicAdd(&ssum[oc], sc * fs);
        atomicAdd(&ssq[oc], sc * sc * fq);
    }
    __syncthreads();
    if (tid < CCH) {
        __hip_atomic_fetch_add(&gsum[tid], ssum[tid], __ATOMIC_RELAXED,
                               __HIP_MEMORY_SCOPE_AGENT);
        __hip_atomic_fetch_add(&gsq[tid], ssq[tid], __ATOMIC_RELAXED,
                               __HIP_MEMORY_SCOPE_AGENT);
    }
}

__global__ void finalize_bn(const float* __restrict__ gsum, const float* __restrict__ gsq,
                            const float* __restrict__ gamma, const float* __restrict__ beta,
                            const float* __restrict__ b1,
                            float* __restrict__ kc, float* __restrict__ dc) {
    int c = threadIdx.x;
    const float inv = 1.0f / (float)NHW;
    float mu  = gsum[c] * inv;
    float var = gsq[c] * inv - mu * mu;
    float rs  = rsqrtf(var + 1e-5f);
    float k   = gamma[c] * rs;
    kc[c] = k;
    dc[c] = beta[c] - mu * k + b1[c];
}

__global__ void __launch_bounds__(256)
conv_final(const unsigned char* __restrict__ abin, const int* __restrict__ bfragG,
           const float* __restrict__ scale, const float* __restrict__ kc,
           const float* __restrict__ dc, const float* __restrict__ x,
           const float* __restrict__ alpha, const float* __restrict__ b2,
           float* __restrict__ out) {
    __shared__ __align__(64) unsigned char smem[36864];
    int* ldsB = (int*)smem;
    float* stageAll = (float*)smem;
    int tid = threadIdx.x;
#pragma unroll
    for (int it = 0; it < 36; ++it) ldsB[tid + it * 256] = bfragG[tid + it * 256];
    __syncthreads();

    int wave = tid >> 5, lane = tid & 31;
    int pair = blockIdx.x * 8 + wave;
    int n0, h0, wb0, n1, h1, wb1;
    decode_tile(pair * 2, n0, h0, wb0);
    decode_tile(pair * 2 + 1, n1, h1, wb1);

    v8i acc[8];
    conv_tile2(abin, ldsB, n0, h0, wb0, n1, h1, wb1, lane, acc);

    int m = lane & 15, half8 = (lane >> 4) & 1;
    float sck[4], scd[4];
#pragma unroll
    for (int g = 0; g < 4; ++g) {
        int oc = g * 16 + m;
        sck[g] = scale[oc] * kc[oc];
        scd[g] = dc[oc];
    }
    __syncthreads();
    float* stage = stageAll + wave * 1024;

#pragma unroll
    for (int t = 0; t < 2; ++t) {
        int n = t ? n1 : n0, h = t ? h1 : h0, wbase = t ? wb1 : wb0;
#pragma unroll
        for (int g = 0; g < 4; ++g) {
            int oc = g * 16 + m;
#pragma unroll
            for (int r = 0; r < 8; ++r) {
                int pix = r + half8 * 8;
                stage[oc * 16 + pix] = (float)acc[t * 4 + g][r] * sck[g] + scd[g];
            }
        }
        size_t obase = (size_t)n * (CCH * HH * WW) + (size_t)h * WW + wbase;
#pragma unroll
        for (int it = 0; it < 8; ++it) {
            int i = lane + it * 32;
            int oc = i >> 2, q = i & 3;
            size_t off = obase + (size_t)oc * (HH * WW) + q * 4;
            float4 xv = *(const float4*)(x + off);
            float4 sv = *(const float4*)(stage + oc * 16 + q * 4);
            float av = alpha[oc], bv = b2[oc];
            float4 o; float tt;
            tt = sv.x + xv.x; o.x = (tt > 0.f ? tt : av * tt) + bv;
            tt = sv.y + xv.y; o.y = (tt > 0.f ? tt : av * tt) + bv;
            tt = sv.z + xv.z; o.z = (tt > 0.f ? tt : av * tt) + bv;
            tt = sv.w + xv.w; o.w = (tt > 0.f ? tt : av * tt) + bv;
            *(float4*)(out + off) = o;
        }
    }
}

extern "C" void kernel_launch(void* const* d_in, const int* in_sizes, int n_in,
                              void* d_out, int out_size, void* d_ws, size_t ws_size,
                              hipStream_t stream) {
    const float* x     = (const float*)d_in[0];
    const float* b0    = (const float*)d_in[1];
    const float* w     = (const float*)d_in[2];
    const float* gamma = (const float*)d_in[3];
    const float* beta  = (const float*)d_in[4];
    const float* b1    = (const float*)d_in[5];
    const float* alpha = (const float*)d_in[6];
    const float* b2    = (const float*)d_in[7];
    float* out = (float*)d_out;

    char* ws = (char*)d_ws;
    float* scale = (float*)(ws + WS_SCALE);
    float* gsum  = (float*)(ws + WS_SUM);
    float* gsq   = (float*)(ws + WS_SQ);
    float* kc    = (float*)(ws + WS_KC);
    float* dc    = (float*)(ws + WS_DC);
    int*   bfrag = (int*)(ws + WS_BFRAG);
    unsigned char* abin = (unsigned char*)(ws + WS_ABIN);

    prep_scale_stats<<<dim3(1), dim3(64), 0, stream>>>(w, scale, gsum, gsq);
    prep_bfrag<<<dim3(36), dim3(32), 0, stream>>>(w, bfrag);
    hipMemsetAsync(abin, 0, ABIN_BYTES, stream);
    binarize_act<<<dim3(NB * HH), dim3(256), 0, stream>>>(x, b0, abin);
    conv_stats<<<dim3(1568), dim3(256), 0, stream>>>(abin, bfrag, scale, gsum, gsq);
    finalize_bn<<<dim3(1), dim3(64), 0, stream>>>(gsum, gsq, gamma, beta, b1, kc, dc);
    conv_final<<<dim3(1568), dim3(256), 0, stream>>>(abin, bfrag, scale, kc, dc,
                                                     x, alpha, b2, out);
}
